// IlluminationLayer_77541339562267
// MI455X (gfx1250) — matrix operand order repaired and checked by reading, not compiled, not run on hardware
//
#include <hip/hip_runtime.h>
#include <hip/hip_bf16.h>
#include <stdint.h>

typedef __attribute__((ext_vector_type(16))) _Float16 v16h;
typedef __attribute__((ext_vector_type(8)))  float    v8f;

#define N_TBINS 1024
#define KOUT    3
#define BATCH   32768

// ---------------------------------------------------------------------------
// Kernel 1: one block, 1024 threads.
//   scaled[n]  = relu(illum[n] * (pc / sum(illum)) + pc/(sbr*N))
//   sqrts[n]   = sqrt(scaled[n])
//   w16[k*N+n] = (f16) cmat_weight[k*N+n]
// Written into d_ws: float scaled[1024]; float sqrts[1024]; _Float16 w16[3072]
// ---------------------------------------------------------------------------
__global__ __launch_bounds__(1024)
void illum_prep_kernel(const float* __restrict__ illum,
                       const float* __restrict__ cmat,
                       const float* __restrict__ photon_count,
                       const float* __restrict__ sbr,
                       float* __restrict__ ws_scaled,   // [1024]
                       float* __restrict__ ws_sqrt,     // [1024]
                       _Float16* __restrict__ ws_w16)   // [3*1024]
{
    __shared__ float red[N_TBINS];
    const int t = threadIdx.x;

    const float il = illum[t];
    red[t] = il;
    __syncthreads();
    for (int s = N_TBINS / 2; s > 0; s >>= 1) {
        if (t < s) red[t] += red[t + s];
        __syncthreads();
    }
    const float area = red[0];

    const float pc  = photon_count[0];
    const float amb = pc / sbr[0] / (float)N_TBINS;
    const float scl = pc / area;

    const float sv = fmaxf(il * scl + amb, 0.0f);   // relu commutes with the gather
    ws_scaled[t] = sv;
    ws_sqrt[t]   = sqrtf(sv);

    for (int i = t; i < KOUT * N_TBINS; i += N_TBINS)
        ws_w16[i] = (_Float16)cmat[i];
}

// ---------------------------------------------------------------------------
// Kernel 2: fused shift -> poisson -> GEMM via v_wmma_f32_16x16x32_f16.
// 256 threads (8 waves) per block, 16 batch rows per wave, 128 rows per block.
// ---------------------------------------------------------------------------
__device__ __forceinline__ uint32_t pcg_hash(uint32_t v) {
    v = v * 747796405u + 2891336453u;
    uint32_t w = ((v >> ((v >> 28u) + 4u)) ^ v) * 277803737u;
    return (w >> 22u) ^ w;
}

#define WROW 1032   // f16 row pitch for LDS weight tile (8-elem pad -> 4-bank skew)

__global__ __launch_bounds__(256)
void illum_wmma_kernel(const int* __restrict__ labels,
                       const float* __restrict__ ws_scaled,
                       const float* __restrict__ ws_sqrt,
                       const _Float16* __restrict__ ws_w16,
                       float* __restrict__ out)        // [BATCH, 3]
{
    __shared__ float    s_scaled[N_TBINS];
    __shared__ float    s_sqrt[N_TBINS];
    __shared__ _Float16 s_w[16 * WROW];   // [16 padded out-cols][1032 K], cols 3..15 zero

    const int tid = threadIdx.x;

    for (int i = tid; i < N_TBINS; i += 256) {
        s_scaled[i] = ws_scaled[i];
        s_sqrt[i]   = ws_sqrt[i];
    }
    for (int i = tid; i < 16 * WROW; i += 256) {
        const int col = i / WROW;
        const int k   = i - col * WROW;
        s_w[i] = (col < KOUT && k < N_TBINS) ? ws_w16[col * N_TBINS + k]
                                             : (_Float16)0.0f;
    }
    __syncthreads();

    const int wave    = tid >> 5;
    const int lane    = tid & 31;
    const int rowBase = blockIdx.x * 128 + wave * 16;
    const int mrow    = lane & 15;           // M within tile for A; N (out col) for B/C/D
    const int hh      = (lane >> 4) & 1;     // half-wave select
    const int row     = rowBase + mrow;
    const int label   = labels[row];

    // B-fragment base: contiguous 16 f16 per lane -> two b128 LDS loads
    const _Float16* wbase = &s_w[mrow * WROW + hh * 16];

    v8f c = {};
    const uint32_t seedBase = (uint32_t)(row << 10);   // row * 1024

    #pragma unroll 1
    for (int chunk = 0; chunk < N_TBINS / 32; ++chunk) {
        const int base = chunk * 32;

        // ---- A fragment: 16 Poisson-sampled f16 values per lane ----
        v16h a;
        #pragma unroll
        for (int e = 0; e < 16; ++e) {
            // A 16x32 layout (contiguous, matches B):
            //  half 0: K = {0..15};  half 1: K = {16..31}
            const int kk  = e + 16 * hh;
            const int bin = base + kk;
            const int src = (bin - label) & (N_TBINS - 1);

            const float lam = s_scaled[src];
            const uint32_t h = pcg_hash(seedBase + (uint32_t)bin);
            const float u1 = (float)(h & 0xffffu) * (1.0f / 65536.0f);
            const float u2 = (float)(h >> 16)     * (1.0f / 65536.0f);
            const float z  = (u1 + u2 - 1.0f) * 2.4494897f;   // approx N(0,1)

            float val = __builtin_fmaf(s_sqrt[src], z, lam);  // normal approx of Poisson
            val = floorf(fmaxf(val, 0.0f) + 0.5f);            // integer count, clamped
            a[e] = (_Float16)val;
        }

        // ---- B fragment: 32x16 f16, K-split across half-waves ----
        v16h b;
        const _Float16* wp = wbase + base;
        #pragma unroll
        for (int e = 0; e < 16; ++e) b[e] = wp[e];

        c = __builtin_amdgcn_wmma_f32_16x16x32_f16(
                false, a, false, b, (short)0, c, false, false);
    }

    // ---- store: C/D layout -> lane holds (M = r + 8*hh, N = mrow) ----
    if (mrow < KOUT) {
        #pragma unroll
        for (int r = 0; r < 8; ++r) {
            const int m = r + 8 * hh;
            out[(rowBase + m) * KOUT + mrow] = c[r];
        }
    }
}

// ---------------------------------------------------------------------------
extern "C" void kernel_launch(void* const* d_in, const int* in_sizes, int n_in,
                              void* d_out, int out_size, void* d_ws, size_t ws_size,
                              hipStream_t stream) {
    (void)in_sizes; (void)n_in; (void)out_size; (void)ws_size;

    const float* illum = (const float*)d_in[0];   // [1024,1]
    const float* cmat  = (const float*)d_in[1];   // [3,1024]
    const int*   lab   = (const int*)d_in[2];     // [32768]
    const float* pc    = (const float*)d_in[3];   // [1]
    const float* sbr   = (const float*)d_in[4];   // [1]
    float*       out   = (float*)d_out;           // [32768,3,1]

    float*    ws_f   = (float*)d_ws;
    float*    scaled = ws_f;                       // [1024]
    float*    sqrts  = ws_f + N_TBINS;             // [1024]
    _Float16* w16    = (_Float16*)(ws_f + 2 * N_TBINS);  // [3*1024]

    illum_prep_kernel<<<1, N_TBINS, 0, stream>>>(illum, cmat, pc, sbr,
                                                 scaled, sqrts, w16);

    // 32768 rows / (8 waves * 16 rows) = 256 blocks
    illum_wmma_kernel<<<BATCH / 128, 256, 0, stream>>>(lab, scaled, sqrts, w16, out);
}
